// GroupedQueryAttention_48498770706470
// MI455X (gfx1250) — matrix operand order repaired and checked by reading, not compiled, not run on hardware
//
#include <hip/hip_runtime.h>

typedef __bf16 bf16;
typedef __attribute__((ext_vector_type(16))) __bf16 v16bf;
typedef __attribute__((ext_vector_type(8)))  __bf16 v8bf;
typedef __attribute__((ext_vector_type(8)))  float  v8f;
typedef __attribute__((ext_vector_type(4))) unsigned int u32x4;
typedef __attribute__((ext_vector_type(8))) int i32x8;
typedef __attribute__((ext_vector_type(4))) int i32x4;

#define B_  2
#define T_  2048
#define D_  2048
#define H_  32
#define G_  8
#define DH_ 64
#define GS_ 4
#define BK_ 64   // K-panel staged in LDS per double-buffer slot

static __device__ __forceinline__ v16bf cat8(v8bf lo, v8bf hi) {
  return __builtin_shufflevector(lo, hi, 0,1,2,3,4,5,6,7,8,9,10,11,12,13,14,15);
}
static __device__ __forceinline__ v8f wmma_bf(v16bf a, v16bf b, v8f c) {
  // D = A(16x32 bf16) * B(32x16 bf16) + C(16x16 f32)
  return __builtin_amdgcn_wmma_f32_16x16x32_bf16(false, a, false, b, (short)0, c, false, false);
}

// ---------------- Tensor Data Mover (CDNA5 TDM) ----------------
#if defined(__AMDGCN__) && __has_builtin(__builtin_amdgcn_tensor_load_to_lds) && \
    __has_builtin(__builtin_amdgcn_s_wait_tensorcnt)
#define HAVE_TDM 1
#endif

// Compile-time probes: visible in hipcc stderr, per compilation pass.
#ifdef __AMDGCN__
#ifdef HAVE_TDM
#if __clang_major__ >= 23
#pragma message("CDNA5-PROBE: device pass, TDM path ENABLED (6-arg tensor_load_to_lds)")
#else
#pragma message("CDNA5-PROBE: device pass, TDM path ENABLED (5-arg tensor_load_to_lds)")
#endif
#else
#pragma message("CDNA5-PROBE: device pass, TDM builtin ABSENT -> cooperative-copy fallback")
#endif
#endif

#ifdef HAVE_TDM
// Issue one 2-D tile DMA: tile_h rows x tile_w bf16 elements, row stride
// row_stride_elems, from gptr into LDS byte offset lds_off (rows packed).
// D# bitfields per CDNA5 ISA ch.8 (group0: count/lds/global/type; group1:
// data_size, tensor dims, tile dims, dim0 stride).
static __device__ __forceinline__ void tdm_load_2d(const void* gptr, unsigned lds_off,
                                                   unsigned tile_w, unsigned tile_h,
                                                   unsigned row_stride_elems) {
  unsigned long long ga = (unsigned long long)(size_t)gptr;
  u32x4 g0;
  g0.x = 1u;                                                  // count=1 (valid user D#)
  g0.y = lds_off;                                             // lds_addr (bytes)
  g0.z = (unsigned)ga;                                        // global_addr[31:0]
  g0.w = (unsigned)((ga >> 32) & 0x01FFFFFFu) | (2u << 30);   // global_addr[56:32] | type=2
  const unsigned td0 = row_stride_elems;                      // tensor_dim0 (no row clip)
  const unsigned td1 = 0x00100000u;                           // tensor_dim1 (large)
  i32x8 g1;
  g1[0] = (int)(1u << 16);                                    // data_size=1 -> 2-byte elems
  g1[1] = (int)((td0 & 0xFFFFu) << 16);                       // tensor_dim0[15:0]
  g1[2] = (int)((td0 >> 16) | ((td1 & 0xFFFFu) << 16));       // dim0[31:16] | dim1[15:0]
  g1[3] = (int)((td1 >> 16) | (tile_w << 16));                // dim1[31:16] | tile_dim0
  g1[4] = (int)(tile_h & 0xFFFFu);                            // tile_dim1 (tile_dim2=0)
  g1[5] = (int)row_stride_elems;                              // tensor_dim0_stride[31:0]
  g1[6] = 0;
  g1[7] = 0;
  i32x4 gz = {0, 0, 0, 0};
#if __clang_major__ >= 23
  i32x8 gz8 = {0, 0, 0, 0, 0, 0, 0, 0};
  __builtin_amdgcn_tensor_load_to_lds(g0, g1, gz, gz, gz8, 0);
#else
  __builtin_amdgcn_tensor_load_to_lds(g0, g1, gz, gz, 0);
#endif
}
#endif

// Stage a 128xBK_ A panel and 128xBK_ W panel into LDS (TDM if available,
// cooperative b128 copies otherwise).
static __device__ __forceinline__ void stage_tiles(bf16 (*sA)[BK_], bf16 (*sB)[BK_],
                                                   const bf16* Ablk, const bf16* Wblk,
                                                   int K, int k0, int tid, int wv) {
  (void)tid; (void)wv;
#ifdef HAVE_TDM
  if (wv == 0) {   // one wave issues both DMAs (EXEC ignored for tensor ops)
    tdm_load_2d(Ablk + k0, (unsigned)(size_t)&sA[0][0], BK_, 128, (unsigned)K);
    tdm_load_2d(Wblk + k0, (unsigned)(size_t)&sB[0][0], BK_, 128, (unsigned)K);
  }
#else
  for (int c = tid; c < 1024; c += 256) {          // 1024 x 16B chunks per panel
    int row = c >> 3, off = (c & 7) * 8;
    *(uint4*)&sA[row][off] = *(const uint4*)(Ablk + (size_t)row * K + k0 + off);
    *(uint4*)&sB[row][off] = *(const uint4*)(Wblk + (size_t)row * K + k0 + off);
  }
#endif
}

// ---------------- fp32 -> bf16 convert ----------------
__global__ __launch_bounds__(256) void cvt_f32_bf16(const float* __restrict__ in,
                                                    bf16* __restrict__ out, int n) {
  int i = blockIdx.x * 256 + threadIdx.x;
  if (i < n) out[i] = (bf16)in[i];
}

// ---------------- V: [B,G,T,DH] f32 -> [B,G,DH,T] bf16 ----------------
__global__ __launch_bounds__(256) void transpose_v(const float* __restrict__ in,
                                                   bf16* __restrict__ out, int n) {
  int i = blockIdx.x * 256 + threadIdx.x;
  if (i >= n) return;
  int dh = i & (DH_ - 1);
  int tt = (i >> 6) & (T_ - 1);
  int bg = i >> 17;                      // / (T_*DH_)
  out[((size_t)bg * DH_ + dh) * T_ + tt] = (bf16)in[i];
}

// ---------------- RMSNorm + RoPE: head-major f32 -> head-major bf16 ----------------
// one wave per row of DH=64 (2 elements/lane, wave32)
__global__ __launch_bounds__(256) void rmsnorm_rope(const float* __restrict__ in,
                                                    const float* __restrict__ scale,
                                                    const float* __restrict__ cs,
                                                    const float* __restrict__ sn,
                                                    bf16* __restrict__ out) {
  const int lane = threadIdx.x & 31;
  const int w    = threadIdx.x >> 5;
  const size_t row = (size_t)blockIdx.x * 8 + w;
  const int t = (int)(row & (T_ - 1));
  const float* p = in + row * DH_;
  float x1 = p[lane], x2 = p[lane + 32];
  float ss = x1 * x1 + x2 * x2;
#pragma unroll
  for (int mk = 16; mk >= 1; mk >>= 1) ss += __shfl_xor(ss, mk, 32);
  float inv = rsqrtf(ss * (1.0f / 64.0f) + 1e-6f);
  float n1 = x1 * inv * scale[lane];
  float n2 = x2 * inv * scale[lane + 32];
  float c1 = cs[t * DH_ + lane], c2 = cs[t * DH_ + lane + 32];
  float s1 = sn[t * DH_ + lane], s2 = sn[t * DH_ + lane + 32];
  out[row * DH_ + lane]      = (bf16)(n1 * c1 - n2 * s1);   // x1*cos - x2*sin
  out[row * DH_ + lane + 32] = (bf16)(n2 * c2 + n1 * s2);   // x2*cos + x1*sin
}

// ---------------- bf16 GEMM: C[M,N] = A[M,K] @ W[N,K]^T ----------------
// TDM-fed, double-buffered LDS staging (64 KB/block). Block = 8 waves,
// 128x128 tile; wave = 32x64 (8 WMMA accumulators).
// MODE 0: C row-major f32.  MODE 1: C scattered head-major f32 [b, n/64, t, n%64].
template <int MODE>
__global__ __launch_bounds__(256) void gemm_bf16(const bf16* __restrict__ A,
                                                 const bf16* __restrict__ W,
                                                 float* __restrict__ C,
                                                 int M, int N, int K, int Tdim) {
  __shared__ __align__(32) bf16 sA[2][128][BK_];
  __shared__ __align__(32) bf16 sB[2][128][BK_];
  const int tid  = threadIdx.x;
  const int lane = tid & 31;
  const int wv   = tid >> 5;               // 0..7
  const int l15  = lane & 15;
  const int half = lane >> 4;
  const int wm = (wv >> 1) * 32;           // 0,32,64,96
  const int wn = (wv & 1) * 64;            // 0,64
  const int bm = blockIdx.y * 128;
  const int bn = blockIdx.x * 128;

  v8f zero = {};
  v8f acc[2][4];
#pragma unroll
  for (int mt = 0; mt < 2; ++mt)
#pragma unroll
    for (int nt = 0; nt < 4; ++nt) acc[mt][nt] = zero;

  const bf16* Ablk = A + (size_t)bm * K;
  const bf16* Wblk = W + (size_t)bn * K;

  stage_tiles(sA[0], sB[0], Ablk, Wblk, K, 0, tid, wv);

  for (int k0 = 0; k0 < K; k0 += BK_) {
    const int buf = (k0 / BK_) & 1;
    if (k0 + BK_ < K) {
      stage_tiles(sA[buf ^ 1], sB[buf ^ 1], Ablk, Wblk, K, k0 + BK_, tid, wv);
#ifdef HAVE_TDM
      if (wv == 0) __builtin_amdgcn_s_wait_tensorcnt((short)2);  // current panel done
#endif
    } else {
#ifdef HAVE_TDM
      if (wv == 0) __builtin_amdgcn_s_wait_tensorcnt((short)0);
#endif
    }
    __syncthreads();   // panel visible to all waves

#pragma unroll
    for (int kk = 0; kk < BK_; kk += 32) {
      const bf16* a0p = &sA[buf][wm + l15][kk + half * 16];
      const bf16* a1p = &sA[buf][wm + 16 + l15][kk + half * 16];
      v16bf a0 = cat8(*(const v8bf*)a0p, *(const v8bf*)(a0p + 8));
      v16bf a1 = cat8(*(const v8bf*)a1p, *(const v8bf*)(a1p + 8));
      v16bf b0 = *(const v16bf*)&sB[buf][wn + l15][kk + half * 16];
      v16bf b1 = *(const v16bf*)&sB[buf][wn + 16 + l15][kk + half * 16];
      v16bf b2 = *(const v16bf*)&sB[buf][wn + 32 + l15][kk + half * 16];
      v16bf b3 = *(const v16bf*)&sB[buf][wn + 48 + l15][kk + half * 16];
      acc[0][0] = wmma_bf(a0, b0, acc[0][0]);
      acc[0][1] = wmma_bf(a0, b1, acc[0][1]);
      acc[0][2] = wmma_bf(a0, b2, acc[0][2]);
      acc[0][3] = wmma_bf(a0, b3, acc[0][3]);
      acc[1][0] = wmma_bf(a1, b0, acc[1][0]);
      acc[1][1] = wmma_bf(a1, b1, acc[1][1]);
      acc[1][2] = wmma_bf(a1, b2, acc[1][2]);
      acc[1][3] = wmma_bf(a1, b3, acc[1][3]);
    }
    __syncthreads();   // all reads done before this buffer is re-filled
  }

#pragma unroll
  for (int mt = 0; mt < 2; ++mt)
#pragma unroll
    for (int nt = 0; nt < 4; ++nt)
#pragma unroll
      for (int r = 0; r < 8; ++r) {
        int m = bm + wm + mt * 16 + r + 8 * half;
        int n = bn + wn + nt * 16 + l15;
        float v = acc[mt][nt][r];
        if (MODE == 0) {
          C[(size_t)m * N + n] = v;
        } else {
          int bb = m / Tdim, tt = m % Tdim;
          int hh = n >> 6, dd = n & 63;
          C[(((size_t)bb * (N >> 6) + hh) * Tdim + tt) * 64 + dd] = v;
        }
      }
  (void)M;
}

// ---------------- flash attention: one wave per 16 query rows ----------------
// q: [B,H,T,64] bf16 ; k: [B,G,T,64] bf16 ; vT: [B,G,64,T] bf16 ; ctx: [B,T,H*64] bf16
__global__ __launch_bounds__(128) void flash_attn(const bf16* __restrict__ q,
                                                  const bf16* __restrict__ k,
                                                  const bf16* __restrict__ vT,
                                                  bf16* __restrict__ ctx) {
  __shared__ __align__(16) bf16 sP[4][16][32];   // per-wave P repack tile
  const int lane = threadIdx.x & 31;
  const int w    = threadIdx.x >> 5;
  const int l15  = lane & 15;
  const int half = lane >> 4;

  int wid = blockIdx.x * 4 + w;
  const int QT = T_ / 16;
  const int qt = wid % QT;  wid /= QT;
  const int h  = wid % H_;
  const int b  = wid / H_;
  const int g  = h / GS_;
  const int q0 = qt * 16;

  // Q fragments (16 rows x 64 dh -> two 16x32 A frags), contiguous dh = 16*half + s per lane
  const bf16* qrow = q + ((size_t)(b * H_ + h) * T_ + q0 + l15) * DH_;
  v16bf aq0 = cat8(*(const v8bf*)(qrow + half * 16),      *(const v8bf*)(qrow + 8 + half * 16));
  v16bf aq1 = cat8(*(const v8bf*)(qrow + 32 + half * 16), *(const v8bf*)(qrow + 40 + half * 16));

  v8f zero = {};
  v8f acc0 = zero, acc1 = zero, acc2 = zero, acc3 = zero;
  float rowm[8], rowl[8];
#pragma unroll
  for (int r = 0; r < 8; ++r) { rowm[r] = -1e30f; rowl[r] = 0.0f; }

  const bf16* kbase = k  + (size_t)(b * G_ + g) * T_ * DH_;
  const bf16* vbase = vT + (size_t)(b * G_ + g) * DH_ * T_;
  const int kend = q0 + 16;          // causal limit (exclusive)
  const float SC = 0.125f;           // 1/sqrt(64)

  for (int kb0 = 0; kb0 < kend; kb0 += 32) {
    // K B-frags: B[d][key] = K[key][d] -> per-lane contiguous 16-element runs
    const bf16* kr0 = kbase + (size_t)(kb0 + l15) * DH_ + half * 16;
    const bf16* kr1 = kbase + (size_t)(kb0 + 16 + l15) * DH_ + half * 16;
    v16bf bk00 = *(const v16bf*)(kr0);
    v16bf bk01 = *(const v16bf*)(kr0 + 32);
    v16bf bk10 = *(const v16bf*)(kr1);
    v16bf bk11 = *(const v16bf*)(kr1 + 32);

    v8f s0 = zero, s1 = zero;                 // 16 q x 16 keys each
    s0 = wmma_bf(aq0, bk00, s0);
    s0 = wmma_bf(aq1, bk01, s0);
    s1 = wmma_bf(aq0, bk10, s1);
    s1 = wmma_bf(aq1, bk11, s1);

    // online softmax over the 32-key chunk (rows live per 16-lane half)
#pragma unroll
    for (int r = 0; r < 8; ++r) {
      int qi = q0 + r + 8 * half;
      float x0 = s0[r] * SC;
      float x1 = s1[r] * SC;
      if (kb0 + l15 > qi)      x0 = -1e30f;
      if (kb0 + 16 + l15 > qi) x1 = -1e30f;
      float mx = fmaxf(x0, x1);
#pragma unroll
      for (int mk = 8; mk >= 1; mk >>= 1) mx = fmaxf(mx, __shfl_xor(mx, mk, 32));
      float mnew = fmaxf(rowm[r], mx);
      float corr = __expf(rowm[r] - mnew);
      float p0 = __expf(x0 - mnew);
      float p1 = __expf(x1 - mnew);
      float ps = p0 + p1;
#pragma unroll
      for (int mk = 8; mk >= 1; mk >>= 1) ps += __shfl_xor(ps, mk, 32);
      rowl[r] = rowl[r] * corr + ps;
      rowm[r] = mnew;
      acc0[r] *= corr; acc1[r] *= corr; acc2[r] *= corr; acc3[r] *= corr;
      sP[w][r + 8 * half][l15]      = (bf16)p0;   // C-layout -> LDS
      sP[w][r + 8 * half][16 + l15] = (bf16)p1;
    }
    // read back P in A layout (DS ops in-order within a wave; tile is wave-private)
    v16bf pa = cat8(*(const v8bf*)&sP[w][l15][half * 16],
                    *(const v8bf*)&sP[w][l15][8 + half * 16]);

    // V B-frags from dh-major vT: B[key][dh] contiguous along keys per lane
    const bf16* vr = vbase + (size_t)l15 * T_ + kb0 + half * 16;
    v16bf bv0 = *(const v16bf*)(vr);
    v16bf bv1 = *(const v16bf*)(vr + (size_t)16 * T_);
    v16bf bv2 = *(const v16bf*)(vr + (size_t)32 * T_);
    v16bf bv3 = *(const v16bf*)(vr + (size_t)48 * T_);
    acc0 = wmma_bf(pa, bv0, acc0);
    acc1 = wmma_bf(pa, bv1, acc1);
    acc2 = wmma_bf(pa, bv2, acc2);
    acc3 = wmma_bf(pa, bv3, acc3);
  }

  bf16* crow = ctx + (size_t)b * T_ * (H_ * DH_) + (size_t)h * DH_;
#pragma unroll
  for (int r = 0; r < 8; ++r) {
    int qi = q0 + r + 8 * half;
    float inv = 1.0f / rowl[r];
    bf16* c = crow + (size_t)qi * (H_ * DH_) + l15;
    c[0]  = (bf16)(acc0[r] * inv);
    c[16] = (bf16)(acc1[r] * inv);
    c[32] = (bf16)(acc2[r] * inv);
    c[48] = (bf16)(acc3[r] * inv);
  }
}

extern "C" void kernel_launch(void* const* d_in, const int* in_sizes, int n_in,
                              void* d_out, int out_size, void* d_ws, size_t ws_size,
                              hipStream_t stream) {
  (void)in_sizes; (void)n_in; (void)out_size; (void)ws_size;
  const float* x  = (const float*)d_in[0];
  // d_in[1] = mask (bool): causal mask is computed arithmetically, not read
  const float* cs = (const float*)d_in[2];
  const float* sn = (const float*)d_in[3];
  const float* Wq = (const float*)d_in[4];
  const float* Wk = (const float*)d_in[5];
  const float* Wv = (const float*)d_in[6];
  const float* Wo = (const float*)d_in[7];
  const float* qs = (const float*)d_in[8];
  const float* ks = (const float*)d_in[9];
  float* out = (float*)d_out;

  const int NX  = B_ * T_ * D_;        // 8388608
  const int NWQ = H_ * DH_ * D_;       // 4194304
  const int NWK = G_ * DH_ * D_;       // 1048576
  const int NQ  = B_ * H_ * T_ * DH_;  // 8388608
  const int NK  = B_ * G_ * T_ * DH_;  // 2097152

  char* ws = (char*)d_ws;
  bf16*  xb  = (bf16*)(ws);                                             // 16.78 MB
  bf16*  wqb = (bf16*)(ws + (size_t)NX * 2);                            //  8.39 MB
  bf16*  wkb = (bf16*)(ws + (size_t)NX * 2 + (size_t)NWQ * 2);          //  2.10 MB
  bf16*  wvb = (bf16*)((char*)wkb + (size_t)NWK * 2);                   //  2.10 MB
  bf16*  wob = (bf16*)((char*)wvb + (size_t)NWK * 2);                   //  8.39 MB
  float* qf  = (float*)((char*)wob + (size_t)NWQ * 2);                  // 33.55 MB
  float* kf  = (float*)((char*)qf + (size_t)NQ * 4);                    //  8.39 MB
  float* vf  = (float*)((char*)kf + (size_t)NK * 4);                    //  8.39 MB  (~88 MB peak)
  // dead-region aliases
  bf16* qbb  = xb;          // bf16 Q (post norm+rope), overlays x_bf16
  bf16* kbb  = wqb;         // bf16 K, overlays Wq_bf16
  bf16* vtb  = wkb;         // bf16 V^T, overlays Wk+Wv bf16
  bf16* ctxb = (bf16*)qf;   // bf16 ctx, overlays q f32

  // 1) convert inputs/weights to bf16
  cvt_f32_bf16<<<(NX  + 255) / 256, 256, 0, stream>>>(x,  xb,  NX);
  cvt_f32_bf16<<<(NWQ + 255) / 256, 256, 0, stream>>>(Wq, wqb, NWQ);
  cvt_f32_bf16<<<(NWK + 255) / 256, 256, 0, stream>>>(Wk, wkb, NWK);
  cvt_f32_bf16<<<(NWK + 255) / 256, 256, 0, stream>>>(Wv, wvb, NWK);
  cvt_f32_bf16<<<(NWQ + 255) / 256, 256, 0, stream>>>(Wo, wob, NWQ);

  // 2) QKV projections (TDM-fed WMMA GEMM, head-major f32 out)
  gemm_bf16<1><<<dim3((H_ * DH_) / 128, (B_ * T_) / 128), 256, 0, stream>>>(
      xb, wqb, qf, B_ * T_, H_ * DH_, D_, T_);
  gemm_bf16<1><<<dim3((G_ * DH_) / 128, (B_ * T_) / 128), 256, 0, stream>>>(
      xb, wkb, kf, B_ * T_, G_ * DH_, D_, T_);
  gemm_bf16<1><<<dim3((G_ * DH_) / 128, (B_ * T_) / 128), 256, 0, stream>>>(
      xb, wvb, vf, B_ * T_, G_ * DH_, D_, T_);

  // 3) RMSNorm + RoPE -> bf16 ; V -> dh-major bf16
  rmsnorm_rope<<<(B_ * H_ * T_) / 8, 256, 0, stream>>>(qf, qs, cs, sn, qbb);
  rmsnorm_rope<<<(B_ * G_ * T_) / 8, 256, 0, stream>>>(kf, ks, cs, sn, kbb);
  transpose_v<<<(NK + 255) / 256, 256, 0, stream>>>(vf, vtb, NK);

  // 4) causal flash attention (WMMA for QK^T and PV)
  flash_attn<<<(B_ * H_ * (T_ / 16)) / 4, 128, 0, stream>>>(qbb, kbb, vtb, ctxb);

  // 5) output projection -> f32 row-major d_out
  gemm_bf16<0><<<dim3(D_ / 128, (B_ * T_) / 128), 256, 0, stream>>>(
      ctxb, wob, out, B_ * T_, D_, H_ * DH_, T_);
}
